// MultiHeadAttention_635655159918
// MI455X (gfx1250) — matrix operand order repaired and checked by reading, not compiled, not run on hardware
//
#include <hip/hip_runtime.h>
#include <hip/hip_bf16.h>

typedef __attribute__((ext_vector_type(16))) _Float16 v16h;
typedef __attribute__((ext_vector_type(8)))  _Float16 v8h;
typedef __attribute__((ext_vector_type(8)))  float    v8f;
typedef __attribute__((ext_vector_type(4)))  float    v4f;

#define B_   4
#define S_   1024
#define D_   1024
#define H_   16
#define DK_  64
#define M_   (B_ * S_)   // 4096

__device__ __forceinline__ v16h cat8(v8h lo, v8h hi) {
    return __builtin_shufflevector(lo, hi, 0,1,2,3,4,5,6,7,8,9,10,11,12,13,14,15);
}

__device__ __forceinline__ v8h load_cvt8(const float* p) {
    v4f a = *(const v4f*)p;
    v4f b = *(const v4f*)(p + 4);
    v8h r;
#pragma unroll
    for (int i = 0; i < 4; ++i) { r[i] = (_Float16)a[i]; r[4 + i] = (_Float16)b[i]; }
    return r;
}

__device__ __forceinline__ v8f zero8() {
    v8f z;
#pragma unroll
    for (int i = 0; i < 8; ++i) z[i] = 0.0f;
    return z;
}

__device__ __forceinline__ v8f wmma_f16(v16h a, v16h b, v8f c) {
    return __builtin_amdgcn_wmma_f32_16x16x32_f16(false, a, false, b, (short)0, c, false, false);
}

// Low 32 bits of a generic pointer to __shared__ data are the LDS byte offset
// (generic shared aperture is SHARED_BASE<<32 | lds_offset, per ISA 10.2).
__device__ __forceinline__ unsigned lds_addr(const void* p) {
    return (unsigned)(uintptr_t)p;
}

// CDNA5 async global->LDS copy: each active lane copies 16B from its global
// address to its LDS address. Tracked with ASYNCcnt.
__device__ __forceinline__ void async_copy_b128(unsigned lds, const void* gptr) {
    asm volatile("global_load_async_to_lds_b128 %0, %1, off"
                 :: "v"(lds), "v"((unsigned long long)(uintptr_t)gptr)
                 : "memory");
}
__device__ __forceinline__ void wait_async0() {
    asm volatile("s_wait_asynccnt 0" ::: "memory");
}

// ---------------------------------------------------------------------------
// Kernel 0: f32 -> f16 bulk convert (activations + weights), 8 elems/thread.
// ---------------------------------------------------------------------------
__global__ __launch_bounds__(256) void cvt_kernel(const float* __restrict__ src,
                                                  _Float16* __restrict__ dst) {
    const size_t i = ((size_t)blockIdx.x * 256 + threadIdx.x) * 8;
    *(v8h*)(dst + i) = load_cvt8(src + i);
}

// ---------------------------------------------------------------------------
// Kernel 1: Q/K/V projections.  Y = X @ W^T, X f16 [4096,1024], W f16 [1024,1024].
// grid = (N/64, M/128, 3); block = 128 (4 waves). Wave w owns rows [32w, 32w+32).
// W tile (64x32) async-staged into double-buffered LDS.
// which==0 -> q [BH][S][DK]; ==1 -> k [BH][S][DK]; ==2 -> vT [BH][DK][S]
// ---------------------------------------------------------------------------
__global__ __launch_bounds__(128) void proj_qkv_kernel(
    const _Float16* __restrict__ xq, const _Float16* __restrict__ xk, const _Float16* __restrict__ xv,
    const _Float16* __restrict__ wq, const _Float16* __restrict__ wk, const _Float16* __restrict__ wv,
    _Float16* __restrict__ qf, _Float16* __restrict__ kf, _Float16* __restrict__ vT)
{
    const int which = blockIdx.z;
    const _Float16* X = (which == 0) ? xq : (which == 1) ? xk : xv;
    const _Float16* W = (which == 0) ? wq : (which == 1) ? wk : wv;

    __shared__ _Float16 Wtile[2][64 * 32];   // 2 x 4 KB, double buffered

    const int tid   = threadIdx.x;
    const int wave  = tid >> 5;
    const int lane  = tid & 31;
    const int ln    = lane & 15;
    const int hi    = lane >> 4;
    const int khalf = hi ? 8 : 0;

    const int mbase = blockIdx.y * 128;
    const int nbase = blockIdx.x * 64;
    const int mrow0 = mbase + wave * 32 + ln;

    auto stage = [&](int buf, int kk) {
#pragma unroll
        for (int o = 0; o < 2; ++o) {
            const int fl = tid + o * 128;            // 0..255
            const int r  = fl >> 2;                  // tile row (n-local) 0..63
            const int sg = (fl & 3) * 8;             // 16B segment in k
            async_copy_b128(lds_addr(&Wtile[buf][r * 32 + sg]),
                            W + (size_t)(nbase + r) * D_ + kk + sg);
        }
    };

    v8f c[2][4];
#pragma unroll
    for (int s = 0; s < 2; ++s)
#pragma unroll
        for (int f = 0; f < 4; ++f) c[s][f] = zero8();

    stage(0, 0);
    for (int kk = 0; kk < D_; kk += 32) {
        const int buf = (kk >> 5) & 1;
        wait_async0();
        __syncthreads();                     // W tile `buf` ready for all waves
        if (kk + 32 < D_) stage(buf ^ 1, kk + 32);

        const _Float16* ar0 = X + (size_t)mrow0 * D_ + kk;
        const _Float16* ar1 = ar0 + (size_t)16 * D_;
        v16h a0 = cat8(*(const v8h*)(ar0 + hi * 16), *(const v8h*)(ar0 + hi * 16 + 8));
        v16h a1 = cat8(*(const v8h*)(ar1 + hi * 16), *(const v8h*)(ar1 + hi * 16 + 8));

        // Batch all B-fragment LDS loads, then the WMMA burst (hides DS latency)
        v16h bf[4];
#pragma unroll
        for (int f = 0; f < 4; ++f) {
            const _Float16* brow = &Wtile[buf][(f * 16 + ln) * 32 + hi * 16];
            bf[f] = cat8(*(const v8h*)brow, *(const v8h*)(brow + 8));
        }
#pragma unroll
        for (int f = 0; f < 4; ++f) {
            c[0][f] = wmma_f16(a0, bf[f], c[0][f]);
            c[1][f] = wmma_f16(a1, bf[f], c[1][f]);
        }
        // LDS reads are consumed by the WMMAs (dscnt-waited) before the next
        // iteration's barrier, so one barrier per step suffices.
    }

    const int b = mbase >> 10;   // 128-row tiles never cross a batch
#pragma unroll
    for (int s = 0; s < 2; ++s) {
        const int srow0 = (mbase & (S_ - 1)) + wave * 32 + s * 16 + hi * 8;
#pragma unroll
        for (int f = 0; f < 4; ++f) {
            const int n  = nbase + f * 16 + ln;
            const int h  = n >> 6;
            const int dk = n & 63;
            if (which < 2) {
                _Float16* dst = ((which == 0) ? qf : kf)
                              + ((size_t)(b * H_ + h) * S_ + srow0) * DK_ + dk;
#pragma unroll
                for (int g = 0; g < 8; ++g) dst[g * DK_] = (_Float16)c[s][f][g];
            } else {
                v8h pk;
#pragma unroll
                for (int g = 0; g < 8; ++g) pk[g] = (_Float16)c[s][f][g];
                *(v8h*)(vT + ((size_t)(b * H_ + h) * DK_ + dk) * S_ + srow0) = pk;
            }
        }
    }
}

// ---------------------------------------------------------------------------
// Kernel 2: flash attention. Block = 4 waves sharing one (b,h); wave w owns the
// 16-row q tile qt = blockIdx*4+w. K/V chunks async-staged to LDS, double buffered.
// ---------------------------------------------------------------------------
__global__ __launch_bounds__(128) void attn_kernel(
    const _Float16* __restrict__ qf, const _Float16* __restrict__ kf,
    const _Float16* __restrict__ vT, _Float16* __restrict__ attn)
{
    __shared__ _Float16 Ktile[2][32 * 64];       // keys x dk, 2 x 4 KB
    __shared__ _Float16 Vtile[2][64 * 32];       // dk x keys, 2 x 4 KB
    __shared__ _Float16 ptile_all[4][16 * 32];   // wave-private P tiles

    const int tid   = threadIdx.x;
    const int wave  = tid >> 5;
    const int lane  = tid & 31;
    const int ln    = lane & 15;
    const int hi    = lane >> 4;
    const int khalf = hi ? 8 : 0;
    _Float16* ptile = ptile_all[wave];

    const int wid = blockIdx.x * 4 + wave;   // 0..4095
    const int qt  = wid & 63;
    const int bh  = wid >> 6;                // same for all 4 waves in a block

    const _Float16* qrow  = qf + ((size_t)bh * S_ + qt * 16 + ln) * DK_;
    const _Float16* kbase = kf + (size_t)bh * S_ * DK_;
    const _Float16* vbase = vT + (size_t)bh * DK_ * S_;

    auto stage = [&](int buf, int j) {
#pragma unroll
        for (int o = 0; o < 2; ++o) {        // K chunk: 32 rows x 128 B
            const int fl = tid + o * 128;
            const int r  = fl >> 3, sg = (fl & 7) * 8;
            async_copy_b128(lds_addr(&Ktile[buf][r * 64 + sg]),
                            kbase + (size_t)(j + r) * DK_ + sg);
        }
#pragma unroll
        for (int o = 0; o < 2; ++o) {        // V chunk: 64 rows x 64 B
            const int fl = tid + o * 128;
            const int r  = fl >> 2, sg = (fl & 3) * 8;
            async_copy_b128(lds_addr(&Vtile[buf][r * 32 + sg]),
                            vbase + (size_t)r * S_ + j + sg);
        }
    };

    // Persistent q A-fragments, pre-scaled by 1/sqrt(DK) = 0.125
    v16h aq[2];
#pragma unroll
    for (int ka = 0; ka < 2; ++ka) {
        v16h t = cat8(*(const v8h*)(qrow + ka * 32 + hi * 16),
                      *(const v8h*)(qrow + ka * 32 + hi * 16 + 8));
#pragma unroll
        for (int i = 0; i < 16; ++i) t[i] = t[i] * (_Float16)0.125f;
        aq[ka] = t;
    }

    float mrun[8], lrun[8];
    v8f acc[4];
#pragma unroll
    for (int g = 0; g < 8; ++g) { mrun[g] = -1e30f; lrun[g] = 0.0f; }
#pragma unroll
    for (int f = 0; f < 4; ++f) acc[f] = zero8();

    stage(0, 0);
    for (int j = 0; j < S_; j += 32) {
        const int buf = (j >> 5) & 1;
        wait_async0();
        __syncthreads();                     // K/V chunk `buf` ready
        if (j + 32 < S_) stage(buf ^ 1, j + 32);

        // ---- scores: S = q @ k^T; batch all four K-fragments first ----
        v16h bk[2][2];   // [key-frag][ka]
#pragma unroll
        for (int ka = 0; ka < 2; ++ka) {
            const _Float16* k0 = &Ktile[buf][(ln)      * 64 + ka * 32 + hi * 16];
            const _Float16* k1 = &Ktile[buf][(16 + ln) * 64 + ka * 32 + hi * 16];
            bk[0][ka] = cat8(*(const v8h*)k0, *(const v8h*)(k0 + 8));
            bk[1][ka] = cat8(*(const v8h*)k1, *(const v8h*)(k1 + 8));
        }
        v8f s0 = zero8(), s1 = zero8();
        s0 = wmma_f16(aq[0], bk[0][0], s0);
        s1 = wmma_f16(aq[0], bk[1][0], s1);
        s0 = wmma_f16(aq[1], bk[0][1], s0);
        s1 = wmma_f16(aq[1], bk[1][1], s1);

        // ---- online softmax (rows live in 16-lane halves; masks 1/2/4/8) ----
#pragma unroll
        for (int g = 0; g < 8; ++g) {
            float vmax = fmaxf(s0[g], s1[g]);
            vmax = fmaxf(vmax, __shfl_xor(vmax, 1));
            vmax = fmaxf(vmax, __shfl_xor(vmax, 2));
            vmax = fmaxf(vmax, __shfl_xor(vmax, 4));
            vmax = fmaxf(vmax, __shfl_xor(vmax, 8));
            const float mnew = fmaxf(mrun[g], vmax);
            const float corr = __expf(mrun[g] - mnew);
            mrun[g] = mnew;
            const float p0 = __expf(s0[g] - mnew);
            const float p1 = __expf(s1[g] - mnew);
            float rs = p0 + p1;
            rs += __shfl_xor(rs, 1);
            rs += __shfl_xor(rs, 2);
            rs += __shfl_xor(rs, 4);
            rs += __shfl_xor(rs, 8);
            lrun[g] = lrun[g] * corr + rs;
#pragma unroll
            for (int f = 0; f < 4; ++f) acc[f][g] *= corr;
            const int row = g + hi * 8;
            ptile[row * 32 + ln]      = (_Float16)p0;
            ptile[row * 32 + 16 + ln] = (_Float16)p1;
        }
        // C-layout -> A-layout transpose via wave-private LDS (DS in-order per wave)
        asm volatile("s_wait_dscnt 0" ::: "memory");

        const _Float16* prow = &ptile[ln * 32 + hi * 16];
        v16h ap = cat8(*(const v8h*)prow, *(const v8h*)(prow + 8));

        // ---- PV: batch the four V B-fragments, then the WMMA burst ----
        v16h bv[4];
#pragma unroll
        for (int f = 0; f < 4; ++f) {
            const _Float16* vr = &Vtile[buf][(f * 16 + ln) * 32 + hi * 16];
            bv[f] = cat8(*(const v8h*)vr, *(const v8h*)(vr + 8));
        }
#pragma unroll
        for (int f = 0; f < 4; ++f) acc[f] = wmma_f16(ap, bv[f], acc[f]);

        asm volatile("" ::: "memory");   // keep next iter's P stores after this iter's loads
    }

    // ---- finalize: divide by row sums, store f16 [row, h*64+dk] ----
    const int b = bh >> 4, h = bh & 15;
#pragma unroll
    for (int g = 0; g < 8; ++g) {
        const float inv = 1.0f / lrun[g];
        const size_t m = (size_t)b * S_ + qt * 16 + hi * 8 + g;
#pragma unroll
        for (int f = 0; f < 4; ++f)
            attn[m * D_ + h * 64 + f * 16 + ln] = (_Float16)(acc[f][g] * inv);
    }
}

// ---------------------------------------------------------------------------
// Kernel 3: output projection + residual.  y = attn @ Wo^T + residual (f32)
// grid = (16, 32); block = 128; wave owns 32x64; Wo tile async double-buffered.
// ---------------------------------------------------------------------------
__global__ __launch_bounds__(128) void oproj_kernel(
    const _Float16* __restrict__ attn, const _Float16* __restrict__ who,
    const float* __restrict__ resid, float* __restrict__ y)
{
    __shared__ _Float16 Wtile[2][64 * 32];

    const int tid   = threadIdx.x;
    const int wave  = tid >> 5;
    const int lane  = tid & 31;
    const int ln    = lane & 15;
    const int hi    = lane >> 4;
    const int khalf = hi ? 8 : 0;

    const int mbase = blockIdx.y * 128;
    const int nbase = blockIdx.x * 64;
    const int mrow0 = mbase + wave * 32 + ln;

    auto stage = [&](int buf, int kk) {
#pragma unroll
        for (int o = 0; o < 2; ++o) {
            const int fl = tid + o * 128;
            const int r  = fl >> 2, sg = (fl & 3) * 8;
            async_copy_b128(lds_addr(&Wtile[buf][r * 32 + sg]),
                            who + (size_t)(nbase + r) * D_ + kk + sg);
        }
    };

    v8f c[2][4];
#pragma unroll
    for (int s = 0; s < 2; ++s)
#pragma unroll
        for (int f = 0; f < 4; ++f) c[s][f] = zero8();

    stage(0, 0);
    for (int kk = 0; kk < D_; kk += 32) {
        const int buf = (kk >> 5) & 1;
        wait_async0();
        __syncthreads();
        if (kk + 32 < D_) stage(buf ^ 1, kk + 32);

        const _Float16* ar0 = attn + (size_t)mrow0 * D_ + kk;
        const _Float16* ar1 = ar0 + (size_t)16 * D_;
        v16h a0 = cat8(*(const v8h*)(ar0 + hi * 16), *(const v8h*)(ar0 + hi * 16 + 8));
        v16h a1 = cat8(*(const v8h*)(ar1 + hi * 16), *(const v8h*)(ar1 + hi * 16 + 8));

        v16h bf[4];
#pragma unroll
        for (int f = 0; f < 4; ++f) {
            const _Float16* brow = &Wtile[buf][(f * 16 + ln) * 32 + hi * 16];
            bf[f] = cat8(*(const v8h*)brow, *(const v8h*)(brow + 8));
        }
#pragma unroll
        for (int f = 0; f < 4; ++f) {
            c[0][f] = wmma_f16(a0, bf[f], c[0][f]);
            c[1][f] = wmma_f16(a1, bf[f], c[1][f]);
        }
    }

#pragma unroll
    for (int s = 0; s < 2; ++s) {
#pragma unroll
        for (int f = 0; f < 4; ++f) {
            const int n = nbase + f * 16 + ln;
#pragma unroll
            for (int g = 0; g < 8; ++g) {
                const size_t m = (size_t)mbase + wave * 32 + s * 16 + hi * 8 + g;
                y[m * D_ + n] = c[s][f][g] + resid[m * D_ + n];
            }
        }
    }
}

// ---------------------------------------------------------------------------
// Kernel 4: LayerNorm over last dim (1024). One 256-thread block per row.
// ---------------------------------------------------------------------------
__global__ __launch_bounds__(256) void ln_kernel(
    const float* __restrict__ y, const float* __restrict__ gamma,
    const float* __restrict__ beta, float* __restrict__ out)
{
    const int row = blockIdx.x;
    const int tid = threadIdx.x;
    const float* x = y + (size_t)row * D_ + tid * 4;

    v4f v = *(const v4f*)x;
    float s  = v[0] + v[1] + v[2] + v[3];
    float s2 = v[0]*v[0] + v[1]*v[1] + v[2]*v[2] + v[3]*v[3];
#pragma unroll
    for (int m = 1; m < 32; m <<= 1) { s += __shfl_xor(s, m); s2 += __shfl_xor(s2, m); }

    __shared__ float redS[8], redS2[8];
    if ((tid & 31) == 0) { redS[tid >> 5] = s; redS2[tid >> 5] = s2; }
    __syncthreads();

    float ts = 0.0f, ts2 = 0.0f;
#pragma unroll
    for (int w = 0; w < 8; ++w) { ts += redS[w]; ts2 += redS2[w]; }

    const float mu   = ts * (1.0f / (float)D_);
    const float var  = ts2 * (1.0f / (float)D_) - mu * mu;
    const float rstd = rsqrtf(var + 1e-6f);

    v4f g = *(const v4f*)(gamma + tid * 4);
    v4f b = *(const v4f*)(beta  + tid * 4);
    v4f o;
#pragma unroll
    for (int i = 0; i < 4; ++i) o[i] = (v[i] - mu) * rstd * g[i] + b[i];
    *(v4f*)(out + (size_t)row * D_ + tid * 4) = o;
}

// ---------------------------------------------------------------------------
extern "C" void kernel_launch(void* const* d_in, const int* in_sizes, int n_in,
                              void* d_out, int out_size, void* d_ws, size_t ws_size,
                              hipStream_t stream) {
    const float* query = (const float*)d_in[0];
    const float* key   = (const float*)d_in[1];
    const float* value = (const float*)d_in[2];
    const float* Wq    = (const float*)d_in[3];
    const float* Wk    = (const float*)d_in[4];
    const float* Wv    = (const float*)d_in[5];
    const float* Wo    = (const float*)d_in[6];
    const float* gamma = (const float*)d_in[7];
    const float* beta  = (const float*)d_in[8];

    char* ws = (char*)d_ws;
    const size_t MB = 1048576;
    _Float16* xq  = (_Float16*)(ws + 0 * MB);    // 8 MiB
    _Float16* xk  = (_Float16*)(ws + 8 * MB);    // 8 MiB
    _Float16* xv  = (_Float16*)(ws + 16 * MB);   // 8 MiB
    _Float16* whq = (_Float16*)(ws + 24 * MB);   // 2 MiB
    _Float16* whk = (_Float16*)(ws + 26 * MB);   // 2 MiB
    _Float16* whv = (_Float16*)(ws + 28 * MB);   // 2 MiB
    _Float16* who = (_Float16*)(ws + 30 * MB);   // 2 MiB
    _Float16* qf  = (_Float16*)(ws + 32 * MB);   // 8 MiB
    _Float16* kf  = (_Float16*)(ws + 40 * MB);   // 8 MiB
    _Float16* vT  = (_Float16*)(ws + 48 * MB);   // 8 MiB
    _Float16* at  = (_Float16*)(ws + 56 * MB);   // 8 MiB
    float*    y   = (float*)   (ws + 0 * MB);    // 16 MiB, aliases dead xq/xk

    // f32 -> f16 conversions (activations: 4M elems; weights: 1M elems)
    cvt_kernel<<<dim3(2048), 256, 0, stream>>>(query, xq);
    cvt_kernel<<<dim3(2048), 256, 0, stream>>>(key,   xk);
    cvt_kernel<<<dim3(2048), 256, 0, stream>>>(value, xv);
    cvt_kernel<<<dim3(512),  256, 0, stream>>>(Wq, whq);
    cvt_kernel<<<dim3(512),  256, 0, stream>>>(Wk, whk);
    cvt_kernel<<<dim3(512),  256, 0, stream>>>(Wv, whv);
    cvt_kernel<<<dim3(512),  256, 0, stream>>>(Wo, who);

    proj_qkv_kernel<<<dim3(D_ / 64, M_ / 128, 3), 128, 0, stream>>>(
        xq, xk, xv, whq, whk, whv, qf, kf, vT);
    attn_kernel<<<dim3((B_ * H_ * (S_ / 16)) / 4), 128, 0, stream>>>(qf, kf, vT, at);
    oproj_kernel<<<dim3(D_ / 64, M_ / 128), 128, 0, stream>>>(at, who, query, y);
    ln_kernel<<<dim3(M_), 256, 0, stream>>>(y, gamma, beta, (float*)d_out);
}
